// PDI_Layer_21053929685388
// MI455X (gfx1250) — matrix operand order repaired and checked by reading, not compiled, not run on hardware
//
#include <hip/hip_runtime.h>

typedef __attribute__((ext_vector_type(16))) _Float16 v16h;
typedef __attribute__((ext_vector_type(8)))  float    v8f;

#define HWDIM 256
#define HWPIX (HWDIM * HWDIM)

__device__ __forceinline__ unsigned pk2(float lo, float hi) {
    union { _Float16 h[2]; unsigned u; } p;
    p.h[0] = (_Float16)lo; p.h[1] = (_Float16)hi;
    return p.u;
}

// 10 raw invariant channels (already divided by the R0 denominators):
// f[0..2]=inv2, f[3..4]=inv3, f[5..6]=inv4, f[7]=inv5, f[8..9]=inv6
__device__ __forceinline__ void invariants(const float ux[3], const float uy[3],
                                           const float uxx[3], const float uyy[3],
                                           const float uxy[3], float f[10]) {
    float r01 = ux[0]*uy[1] - ux[1]*uy[0];
    float r02 = ux[1]*uy[2] - ux[2]*uy[1];
    float r03 = ux[2]*uy[0] - ux[0]*uy[2];
    float R0  = (r01 + r02 + r03) * (1.0f / 3.0f);
    float i234 = 1.0f / (1.0f + R0 * R0);
    float den56 = 1.0f + R0;
    if (den56 == 0.0f) den56 = 1.0f;
    float i56 = 1.0f / den56;
#pragma unroll
    for (int c = 0; c < 3; ++c)
        f[c] = (uxx[c]*uy[c]*uy[c] - 2.0f*uxy[c]*ux[c]*uy[c] + uyy[c]*ux[c]*ux[c]) * i234;
#pragma unroll
    for (int c = 0; c < 2; ++c) {
        float t = 2.0f*(uxx[c]*uy[c]*uy[c+1] - uxy[c]*ux[c]*uy[c+1]
                      - uxy[c]*uy[c]*ux[c+1] + uyy[c]*ux[c]*ux[c+1])
                + (uxx[c+1]*uy[c]*uy[c] - 2.0f*uxy[c+1]*ux[c]*uy[c] + uyy[c+1]*ux[c]*ux[c]);
        f[3 + c] = t * i234;
    }
#pragma unroll
    for (int c = 0; c < 2; ++c) {
        float t = 2.0f*(uxx[c+1]*uy[c+1]*uy[c] - uxy[c+1]*ux[c+1]*uy[c]
                      - uxy[c+1]*uy[c+1]*ux[c] + uyy[c+1]*ux[c+1]*ux[c])
                + (uxx[c]*uy[c+1]*uy[c+1] - 2.0f*uxy[c]*ux[c+1]*uy[c+1] + uyy[c]*ux[c+1]*ux[c+1]);
        f[5 + c] = t * i234;
    }
    f[7] = (ux[0]*uy[2] - uy[0]*ux[2]) * i56;
#pragma unroll
    for (int c = 0; c < 2; ++c)
        f[8 + c] = (ux[c]*uy[c+1] - uy[c]*ux[c+1]) * i56;
}

__global__ void init_max(unsigned* maxbuf) {
    if (threadIdx.x < 48) maxbuf[threadIdx.x] = 0u;
}

// ---------------- Pass A: per-batch max-abs of |x|, |inv234|, |inv56| ----------------
__global__ __launch_bounds__(256) void pass_max(const float* __restrict__ x,
                                                const float* __restrict__ kd,
                                                unsigned* __restrict__ maxbuf) {
    __shared__ float kds[405];
    __shared__ float tile[3][24][24];
    __shared__ float red[256];

    int blk = blockIdx.x;
    int n = blk >> 8;          // 256 tiles per batch image
    int t = blk & 255;
    int y0 = (t >> 4) * 16, x0 = (t & 15) * 16;
    int tid = threadIdx.x;

    for (int i = tid; i < 405; i += 256) kds[i] = kd[i];
    for (int i = tid; i < 3 * 24 * 24; i += 256) {
        int c = i / 576, r = i % 576;
        int iy = r / 24, ix = r % 24;
        int gy = y0 + iy - 4, gx = x0 + ix - 4;
        float v = 0.0f;
        if (gy >= 0 && gy < HWDIM && gx >= 0 && gx < HWDIM)
            v = x[((size_t)n * 3 + c) * HWPIX + (size_t)gy * HWDIM + gx];
        tile[c][iy][ix] = v;
    }
    __syncthreads();

    int ly = tid >> 4, lx = tid & 15;
    float ux[3], uy[3], uxx[3], uyy[3], uxy[3];
    float mx = 0.0f;
#pragma unroll
    for (int c = 0; c < 3; ++c) {
        float s0 = 0, s1 = 0, s2 = 0, s3 = 0, s4 = 0;
        for (int iy = 0; iy < 9; ++iy)
#pragma unroll
            for (int ix = 0; ix < 9; ++ix) {
                float v = tile[c][ly + iy][lx + ix];
                int kk = iy * 9 + ix;
                s0 = fmaf(kds[kk], v, s0);
                s1 = fmaf(kds[81 + kk], v, s1);
                s2 = fmaf(kds[162 + kk], v, s2);
                s3 = fmaf(kds[243 + kk], v, s3);
                s4 = fmaf(kds[324 + kk], v, s4);
            }
        ux[c] = s0; uy[c] = s1; uxx[c] = s2; uyy[c] = s3; uxy[c] = s4;
        mx = fmaxf(mx, fabsf(tile[c][ly + 4][lx + 4]));
    }
    float f[10];
    invariants(ux, uy, uxx, uyy, uxy, f);
    float m234 = 0.0f, m56 = 0.0f;
#pragma unroll
    for (int i = 0; i < 7; ++i) m234 = fmaxf(m234, fabsf(f[i]));
#pragma unroll
    for (int i = 7; i < 10; ++i) m56 = fmaxf(m56, fabsf(f[i]));

    float vals[3] = { mx, m234, m56 };
    for (int i = 0; i < 3; ++i) {
        red[tid] = vals[i];
        __syncthreads();
        for (int s = 128; s > 0; s >>= 1) {
            if (tid < s) red[tid] = fmaxf(red[tid], red[tid + s]);
            __syncthreads();
        }
        if (tid == 0) atomicMax(&maxbuf[n * 3 + i], __float_as_uint(red[0]));
        __syncthreads();
    }
}

// ---------------- Pass B: recompute feats, normalize, WMMA MLP ----------------
__global__ __launch_bounds__(256) void pass_mlp(const float* __restrict__ x,
                                                const float* __restrict__ kd,
                                                const float* __restrict__ w1,
                                                const float* __restrict__ b1,
                                                const float* __restrict__ w2,
                                                const float* __restrict__ b2,
                                                const unsigned* __restrict__ maxbuf,
                                                float* __restrict__ out) {
    __shared__ float    kds[405];
    __shared__ float    tile[3][16][24];      // 8-row x 16-col pixel tile + 4 halo
    __shared__ unsigned featT2[16][132];      // half2: featsT [K/2=16][128 pixels]
    __shared__ unsigned w1p[64][17];          // half2: w1 [64 hid][K/2 pairs]
    __shared__ unsigned w2p[64][33];          // half2: w2 [64 out][K/2 pairs]
    __shared__ unsigned HT2[32][132];         // half2: hiddenT [64/2][128 pixels]
    __shared__ float    b1s[64], b2s[64];

    int blk = blockIdx.x;
    int n = blk >> 9;          // 512 tiles per batch image
    int t = blk & 511;
    int y0 = (t >> 4) * 8, x0 = (t & 15) * 16;
    int tid = threadIdx.x;

    float sx = __uint_as_float(maxbuf[n * 3 + 0]); if (sx == 0.0f) sx = 1.0f;
    float s234 = __uint_as_float(maxbuf[n * 3 + 1]); if (s234 == 0.0f) s234 = 1.0f;
    float s56 = __uint_as_float(maxbuf[n * 3 + 2]); if (s56 == 0.0f) s56 = 1.0f;
    float rsx = 1.0f / sx, rs234 = 1.0f / s234, rs56 = 1.0f / s56;

    for (int i = tid; i < 405; i += 256) kds[i] = kd[i];
    for (int i = tid; i < 3 * 16 * 24; i += 256) {
        int c = i / 384, r = i % 384;
        int iy = r / 24, ix = r % 24;
        int gy = y0 + iy - 4, gx = x0 + ix - 4;
        float v = 0.0f;
        if (gy >= 0 && gy < HWDIM && gx >= 0 && gx < HWDIM)
            v = x[((size_t)n * 3 + c) * HWPIX + (size_t)gy * HWDIM + gx];
        tile[c][iy][ix] = v;
    }
    for (int i = tid; i < 64 * 16; i += 256) {        // w1: (64,13) -> padded K=32 pairs
        int h = i >> 4, kk = i & 15;
        float lo = (2 * kk     < 13) ? w1[h * 13 + 2 * kk]     : 0.0f;
        float hi = (2 * kk + 1 < 13) ? w1[h * 13 + 2 * kk + 1] : 0.0f;
        w1p[h][kk] = pk2(lo, hi);
    }
    for (int i = tid; i < 64 * 32; i += 256) {        // w2: (64,64)
        int h = i >> 5, kk = i & 31;
        w2p[h][kk] = pk2(w2[h * 64 + 2 * kk], w2[h * 64 + 2 * kk + 1]);
    }
    if (tid < 64) { b1s[tid] = b1[tid]; b2s[tid] = b2[tid]; }
    __syncthreads();

    // ---- feature construction: threads 0..127 each own one pixel of the 16x8 tile
    if (tid < 128) {
        int ly = tid >> 4, lx = tid & 15;
        float ux[3], uy[3], uxx[3], uyy[3], uxy[3];
#pragma unroll
        for (int c = 0; c < 3; ++c) {
            float s0 = 0, s1 = 0, s2 = 0, s3 = 0, s4 = 0;
            for (int iy = 0; iy < 9; ++iy)
#pragma unroll
                for (int ix = 0; ix < 9; ++ix) {
                    float v = tile[c][ly + iy][lx + ix];
                    int kk = iy * 9 + ix;
                    s0 = fmaf(kds[kk], v, s0);
                    s1 = fmaf(kds[81 + kk], v, s1);
                    s2 = fmaf(kds[162 + kk], v, s2);
                    s3 = fmaf(kds[243 + kk], v, s3);
                    s4 = fmaf(kds[324 + kk], v, s4);
                }
            ux[c] = s0; uy[c] = s1; uxx[c] = s2; uyy[c] = s3; uxy[c] = s4;
        }
        float f[10];
        invariants(ux, uy, uxx, uyy, uxy, f);
        float ft[13];
#pragma unroll
        for (int c = 0; c < 3; ++c) ft[c] = tile[c][ly + 4][lx + 4] * rsx;
#pragma unroll
        for (int i = 0; i < 7; ++i) ft[3 + i] = f[i] * rs234;
#pragma unroll
        for (int i = 0; i < 3; ++i) ft[10 + i] = f[7 + i] * rs56;
#pragma unroll
        for (int kk = 0; kk < 8; ++kk) {
            float lo = (2 * kk     < 13) ? ft[2 * kk]     : 0.0f;
            float hi = (2 * kk + 1 < 13) ? ft[2 * kk + 1] : 0.0f;
            featT2[kk][tid] = pk2(lo, hi);
        }
#pragma unroll
        for (int kk = 8; kk < 16; ++kk) featT2[kk][tid] = 0u;
    }
    __syncthreads();

    // ---- GEMMs: 8 waves, wave = pixel tile (16 pixels); loop over 16-wide M tiles
    int wave = tid >> 5, lane = tid & 31;
    int g = lane >> 4, lm = lane & 15;
    int pt = wave;                                    // pixel tile 0..7

    // Layer 1: D[hid=16, pix=16] = w1(16x32) * featsT(32x16); relu -> HT2
#pragma unroll
    for (int mt = 0; mt < 4; ++mt) {
        union { v16h v; unsigned u[8]; } A, B;
#pragma unroll
        for (int r = 0; r < 8; ++r) {
            int kk = (r >> 2) * 8 + g * 4 + (r & 3);  // A 16-bit layout: k = (r/4)*16 + g*8 + 2*(r%4)
            A.u[r] = w1p[mt * 16 + lm][kk];
            B.u[r] = featT2[kk][pt * 16 + lm];        // B 16-bit layout: k = (r/4)*16 + g*8 + 2*(r%4)
        }
        v8f c = {};
        c = __builtin_amdgcn_wmma_f32_16x16x32_f16(false, A.v, false, B.v, (short)0, c, false, false);
#pragma unroll
        for (int q = 0; q < 4; ++q) {                 // C: VGPR j -> M = j + 8*g
            int h0 = mt * 16 + 8 * g + 2 * q;
            float lo = fmaxf(c[2 * q]     + b1s[h0],     0.0f);
            float hi = fmaxf(c[2 * q + 1] + b1s[h0 + 1], 0.0f);
            HT2[mt * 8 + g * 4 + q][pt * 16 + lm] = pk2(lo, hi);
        }
    }
    __syncthreads();

    // Layer 2: D[out=16, pix=16] = w2(16x64) * hiddenT(64x16), K split in two 32-chunks
    size_t outbase = (size_t)n * 64 * HWPIX;
    int prow = y0 + pt;
#pragma unroll
    for (int mt = 0; mt < 4; ++mt) {
        v8f c = {};
#pragma unroll
        for (int ks = 0; ks < 2; ++ks) {
            union { v16h v; unsigned u[8]; } A, B;
#pragma unroll
            for (int r = 0; r < 8; ++r) {
                int kk = ks * 16 + (r >> 2) * 8 + g * 4 + (r & 3);
                A.u[r] = w2p[mt * 16 + lm][kk];
                B.u[r] = HT2[kk][pt * 16 + lm];
            }
            c = __builtin_amdgcn_wmma_f32_16x16x32_f16(false, A.v, false, B.v, (short)0, c, false, false);
        }
#pragma unroll
        for (int j = 0; j < 8; ++j) {
            int o = mt * 16 + j + 8 * g;
            out[outbase + (size_t)o * HWPIX + (size_t)prow * HWDIM + x0 + lm] = c[j] + b2s[o];
        }
    }
}

extern "C" void kernel_launch(void* const* d_in, const int* in_sizes, int n_in,
                              void* d_out, int out_size, void* d_ws, size_t ws_size,
                              hipStream_t stream) {
    (void)in_sizes; (void)n_in; (void)out_size; (void)ws_size;
    const float* x  = (const float*)d_in[0];
    const float* kd = (const float*)d_in[1];
    const float* w1 = (const float*)d_in[2];
    const float* b1 = (const float*)d_in[3];
    const float* w2 = (const float*)d_in[4];
    const float* b2 = (const float*)d_in[5];
    float* out = (float*)d_out;
    unsigned* maxbuf = (unsigned*)d_ws;   // 48 uints: [batch][{|x|,|inv234|,|inv56|}]

    init_max<<<1, 64, 0, stream>>>(maxbuf);
    pass_max<<<16 * 256, 256, 0, stream>>>(x, kd, maxbuf);
    pass_mlp<<<16 * 512, 256, 0, stream>>>(x, kd, w1, b1, w2, b2, maxbuf, out);
}
